// MoE_30313879175949
// MI455X (gfx1250) — matrix operand order repaired and checked by reading, not compiled, not run on hardware
//
#include <hip/hip_runtime.h>
#include <hip/hip_bf16.h>

// ---------------------------------------------------------------------------
// MoE top-1 FFN for MI455X (gfx1250, wave32, WMMA bf16, async LDS copies).
//   D_MODEL=1024, D_HIDDEN=4096, E=8, tokens = B*T = 4096.
// ---------------------------------------------------------------------------

#define NTOK     4096
#define DMODEL   1024
#define DHID     4096
#define NEXP     8

typedef __attribute__((ext_vector_type(16))) __bf16 v16bf;
typedef __attribute__((ext_vector_type(2)))  __bf16 v2bf;
typedef __attribute__((ext_vector_type(8)))  float  v8f;
typedef int v4i __attribute__((vector_size(16)));   // matches builtin param type

union FragAB { v16bf v; uint4 q[2]; };

#ifndef __has_builtin
#define __has_builtin(x) 0
#endif

#if __has_builtin(__builtin_amdgcn_global_load_async_to_lds_b128)
#define HAVE_ASYNC 1
#else
#define HAVE_ASYNC 0
#endif

// Pack two f32 into packed bf16 (RNE). Full 32-bit ops + one v_perm_b32:
// result = { bf16(b) , bf16(a) }  (a in low half).
__device__ __forceinline__ unsigned int pack_bf16(float a, float b) {
#if __has_builtin(__builtin_amdgcn_cvt_pk_bf16_f32)
    v2bf p = __builtin_amdgcn_cvt_pk_bf16_f32(a, b);
    return __builtin_bit_cast(unsigned int, p);
#else
    unsigned int ua = __float_as_uint(a);
    unsigned int ub = __float_as_uint(b);
    ua += 0x7FFFu + ((ua >> 16) & 1u);     // round to nearest even
    ub += 0x7FFFu + ((ub >> 16) & 1u);
    return __builtin_amdgcn_perm(ub, ua, 0x07060302u);  // [ub.hi16 : ua.hi16]
#endif
}

// Async global -> LDS (128-bit per lane), ASYNCcnt-tracked; VGPR-bypassing.
__device__ __forceinline__ void async_copy_b128(const void* g, void* l) {
#if HAVE_ASYNC
    __builtin_amdgcn_global_load_async_to_lds_b128(
        (__attribute__((address_space(1))) v4i*)g,
        (__attribute__((address_space(3))) v4i*)l, 0, 0);
#else
    *(uint4*)l = *(const uint4*)g;
#endif
}

__device__ __forceinline__ void wait_async() {
#if HAVE_ASYNC
#if __has_builtin(__builtin_amdgcn_s_wait_asynccnt)
    __builtin_amdgcn_s_wait_asynccnt(0);
#else
    asm volatile("s_wait_asynccnt 0" ::: "memory");
#endif
#endif
}

// ---------------------------------------------------------------------------
// Gating: one wave per token; top-1 routing into per-expert buckets.
// ---------------------------------------------------------------------------
__global__ __launch_bounds__(256)
void gate_kernel(const float* __restrict__ x, const float* __restrict__ Wg,
                 const float* __restrict__ bg, int* __restrict__ counts,
                 int* __restrict__ buckets, float* __restrict__ wtok) {
    const int wid  = threadIdx.x >> 5;
    const int lane = threadIdx.x & 31;
    const int t    = blockIdx.x * 8 + wid;

    float acc[NEXP];
#pragma unroll
    for (int e = 0; e < NEXP; ++e) acc[e] = 0.f;

    const float* xr = x + (size_t)t * DMODEL;
    for (int d = lane; d < DMODEL; d += 32) {
        const float xv = xr[d];
        const float* wr = Wg + (size_t)d * NEXP;
#pragma unroll
        for (int e = 0; e < NEXP; ++e) acc[e] += xv * wr[e];
    }
#pragma unroll
    for (int e = 0; e < NEXP; ++e) {
#pragma unroll
        for (int off = 16; off > 0; off >>= 1)
            acc[e] += __shfl_xor(acc[e], off, 32);
    }
    if (lane == 0) {
        float m = -1e30f; int bi = 0;
        float lg[NEXP];
#pragma unroll
        for (int e = 0; e < NEXP; ++e) {
            lg[e] = acc[e] + bg[e];
            if (lg[e] > m) { m = lg[e]; bi = e; }
        }
        float denom = 0.f;
#pragma unroll
        for (int e = 0; e < NEXP; ++e) denom += __expf(lg[e] - m);
        const int pos = atomicAdd(&counts[bi], 1);
        buckets[bi * NTOK + pos] = t;
        wtok[t] = 1.0f / denom;
    }
}

// ---------------------------------------------------------------------------
// x fp32 -> bf16 (4096x1024), 4 elems/thread.
// ---------------------------------------------------------------------------
__global__ __launch_bounds__(256)
void cvt_x_kernel(const float* __restrict__ x, unsigned short* __restrict__ xb) {
    const int i = (blockIdx.x * 256 + threadIdx.x) * 4;
    const float4 v = *(const float4*)(x + i);
    *(uint2*)(xb + i) = make_uint2(pack_bf16(v.x, v.y), pack_bf16(v.z, v.w));
}

// ---------------------------------------------------------------------------
// Grouped expert GEMM, 128x128 tile, K-step 32, double-buffered LDS.
// 8 waves (2x4); each wave 64x32 via 4x2 v_wmma_f32_16x16x32_bf16.
// A tile: bf16, gathered by token, async global->LDS copies.
// B tile: fp32 weights, coalesced k-major loads, bf16-packed ds_store_b128.
// ---------------------------------------------------------------------------
#define TM 128
#define TN 128
#define TK 32
#define LDA 40   // ushort stride: 80B = 20 banks -> conflict-free b128 frags
#define LDB 40

template <int K_TOTAL, int N_TOTAL, bool SECOND>
__global__ __launch_bounds__(256)
void ffn_gemm(const unsigned short* __restrict__ Abase, // bf16, row = token id
              const float* __restrict__ Wall,           // (E, K_TOTAL, N_TOTAL)
              const float* __restrict__ bias,           // (E, N_TOTAL)
              const int* __restrict__ counts,
              const int* __restrict__ buckets,
              const float* __restrict__ wtok,
              unsigned short* __restrict__ Hout,        // (NTOK, DHID) bf16
              float* __restrict__ Yout) {               // (NTOK, DMODEL) f32
    const int e  = blockIdx.z;
    const int Me = counts[e];
    const int m0 = blockIdx.y * TM;
    if (m0 >= Me) return;
    const int n0 = blockIdx.x * TN;

    __shared__ unsigned short sA[2][TM * LDA];
    __shared__ unsigned short sB[2][TN * LDB];
    __shared__ int   sTok[TM];
    __shared__ float sStage[8 * 16 * 17];

    const int tid  = threadIdx.x;
    const int wid  = tid >> 5;
    const int lane = tid & 31;
    const int wm   = (wid & 1) * 64;
    const int wn   = (wid >> 1) * 32;

    // B staging role: one n-column, 16 k-contiguous values per thread.
    const int bn  = tid & 127;
    const int bkh = tid >> 7;          // 0 or 1 -> k in [bkh*16, bkh*16+16)

    if (tid < TM) {
        const int row = m0 + tid;
        sTok[tid] = (row < Me) ? buckets[e * NTOK + row] : -1;
    }
    __syncthreads();

    const float* W = Wall + (size_t)e * K_TOTAL * N_TOTAL;

    // ---- tile loaders -----------------------------------------------------
    auto loadA = [&](int buf, int k0) {
#pragma unroll
        for (int i = 0; i < 2; ++i) {
            const int c = tid * 2 + i;                 // 512 x b128 chunks
            const int row = c >> 2, seg = c & 3;
            const int tok = sTok[row];
            unsigned short* lp = &sA[buf][row * LDA + seg * 8];
            if (tok >= 0)
                async_copy_b128(Abase + (size_t)tok * K_TOTAL + k0 + seg * 8, lp);
            else
                *(uint4*)lp = make_uint4(0u, 0u, 0u, 0u);
        }
    };
    auto loadB_glb = [&](int k0, float* r) {
        const float* base = W + (size_t)(k0 + bkh * 16) * N_TOTAL + n0 + bn;
#pragma unroll
        for (int kk = 0; kk < 16; ++kk)                // lanes: consecutive n
            r[kk] = base[(size_t)kk * N_TOTAL];
    };
    auto storeB_lds = [&](int buf, const float* r) {
        unsigned int pk[8];
#pragma unroll
        for (int c = 0; c < 8; ++c) pk[c] = pack_bf16(r[2 * c], r[2 * c + 1]);
        uint4* dst = (uint4*)&sB[buf][bn * LDB + bkh * 16];
        dst[0] = make_uint4(pk[0], pk[1], pk[2], pk[3]);
        dst[1] = make_uint4(pk[4], pk[5], pk[6], pk[7]);
    };

    v8f acc[4][2];
#pragma unroll
    for (int i = 0; i < 4; ++i)
#pragma unroll
        for (int j = 0; j < 2; ++j)
#pragma unroll
            for (int t = 0; t < 8; ++t) acc[i][j][t] = 0.f;

    // ---- prologue: tile 0 -------------------------------------------------
    float breg[16];
    loadA(0, 0);
    loadB_glb(0, breg);
    storeB_lds(0, breg);
    wait_async();
    __syncthreads();

    const int nsteps = K_TOTAL / TK;
    for (int s = 0; s < nsteps; ++s) {
        const int cur = s & 1, nxt = cur ^ 1;
        const bool more = (s + 1) < nsteps;

        if (more) {
            loadB_glb((s + 1) * TK, breg);             // issue early
            loadA(nxt, (s + 1) * TK);                  // async into nxt
            if (s + 2 < nsteps)                        // L2 prefetch, 2 ahead
                __builtin_prefetch(W + (size_t)((s + 2) * TK + bkh * 16) * N_TOTAL
                                     + n0 + bn, 0, 1);
        }

        // ---- fragments: contiguous k order (k = 16*(lane/16) + s), same in A and B ----
        FragAB bfrag[2];
#pragma unroll
        for (int j = 0; j < 2; ++j) {
            const int n  = wn + j * 16 + (lane & 15);
            const int kh = (lane >> 4) * 16;           // K = kh .. kh+15
            bfrag[j].q[0] = *(const uint4*)&sB[cur][n * LDB + kh];
            bfrag[j].q[1] = *(const uint4*)&sB[cur][n * LDB + kh + 8];
        }
        FragAB afrag[4];
#pragma unroll
        for (int i = 0; i < 4; ++i) {
            const int r  = wm + i * 16 + (lane & 15);
            const int kh = (lane >> 4) * 16;           // K = kh .. kh+15
            afrag[i].q[0] = *(const uint4*)&sA[cur][r * LDA + kh];
            afrag[i].q[1] = *(const uint4*)&sA[cur][r * LDA + kh + 8];
        }
#pragma unroll
        for (int i = 0; i < 4; ++i)
#pragma unroll
            for (int j = 0; j < 2; ++j)
                acc[i][j] = __builtin_amdgcn_wmma_f32_16x16x32_bf16(
                    false, afrag[i].v, false, bfrag[j].v,
                    (short)0, acc[i][j], false, false);

        if (more) storeB_lds(nxt, breg);               // after wmma issue
        wait_async();
        __syncthreads();
    }

    // ---- epilogue: frag-at-a-time LDS staging, packed global stores -------
    float* slice = sStage + wid * (16 * 17);
    const int col  = lane & 15;
    const int rb   = (lane >> 4) << 3;
    const int r    = lane >> 1;                        // 0..15
    const int half = lane & 1;                         // 8-col half
#pragma unroll
    for (int j = 0; j < 2; ++j) {
        const int ncol = n0 + wn + j * 16 + half * 8;
        float bv[8];
#pragma unroll
        for (int c = 0; c < 8; ++c) bv[c] = bias[(size_t)e * N_TOTAL + ncol + c];
#pragma unroll
        for (int i = 0; i < 4; ++i) {
#pragma unroll
            for (int t = 0; t < 8; ++t)
                slice[(rb + t) * 17 + col] = acc[i][j][t];
            // same-wave LDS ops are in-order: safe to read back immediately
            const int grow = m0 + wm + i * 16 + r;
            if (grow < Me) {
                const int tok = sTok[wm + i * 16 + r];
                float v[8];
#pragma unroll
                for (int c = 0; c < 8; ++c)
                    v[c] = slice[r * 17 + half * 8 + c] + bv[c];
                if (!SECOND) {
                    unsigned int pk[4];
#pragma unroll
                    for (int c = 0; c < 4; ++c) {
                        float a = v[2 * c], b = v[2 * c + 1];
                        a = a / (1.f + __expf(-a));    // silu
                        b = b / (1.f + __expf(-b));
                        pk[c] = pack_bf16(a, b);
                    }
                    *(uint4*)(Hout + (size_t)tok * DHID + ncol) =
                        make_uint4(pk[0], pk[1], pk[2], pk[3]);
                } else {
                    const float w = wtok[tok];
                    float* yp = Yout + (size_t)tok * DMODEL + ncol;
                    *(float4*)(yp)     = make_float4(v[0] * w, v[1] * w, v[2] * w, v[3] * w);
                    *(float4*)(yp + 4) = make_float4(v[4] * w, v[5] * w, v[6] * w, v[7] * w);
                }
            }
        }
    }
}

// ---------------------------------------------------------------------------
extern "C" void kernel_launch(void* const* d_in, const int* in_sizes, int n_in,
                              void* d_out, int out_size, void* d_ws, size_t ws_size,
                              hipStream_t stream) {
    const float* x  = (const float*)d_in[0];
    const float* Wg = (const float*)d_in[1];
    const float* bg = (const float*)d_in[2];
    const float* W1 = (const float*)d_in[3];
    const float* b1 = (const float*)d_in[4];
    const float* W2 = (const float*)d_in[5];
    const float* b2 = (const float*)d_in[6];
    float* y = (float*)d_out;

    char* ws = (char*)d_ws;
    int*   counts  = (int*)ws;                                  // 32B (pad 256)
    int*   buckets = (int*)(ws + 256);                          // 128KB
    float* wtok    = (float*)(ws + 256 + NEXP * NTOK * 4);      // 16KB
    unsigned short* xb = (unsigned short*)((char*)wtok + NTOK * 4);  // 8MB
    unsigned short* h  = xb + (size_t)NTOK * DMODEL;                 // 32MB

    (void)hipMemsetAsync(counts, 0, NEXP * sizeof(int), stream);

    gate_kernel<<<NTOK / 8, 256, 0, stream>>>(x, Wg, bg, counts, buckets, wtok);
    cvt_x_kernel<<<(NTOK * DMODEL) / (256 * 4), 256, 0, stream>>>(x, xb);

    ffn_gemm<DMODEL, DHID, false>
        <<<dim3(DHID / TN, NTOK / TM, NEXP), 256, 0, stream>>>(
            xb, W1, b1, counts, buckets, wtok, h, nullptr);

    ffn_gemm<DHID, DMODEL, true>
        <<<dim3(DMODEL / TN, NTOK / TM, NEXP), 256, 0, stream>>>(
            h, W2, b2, counts, buckets, wtok, nullptr, y);
}
